// MoEGate_52673478918592
// MI455X (gfx1250) — matrix operand order repaired and checked by reading, not compiled, not run on hardware
//
#include <hip/hip_runtime.h>

typedef __attribute__((ext_vector_type(16))) __bf16 v16bf;
typedef __attribute__((ext_vector_type(8)))  float  v8f;

#define H_DIM  7168
#define N_EXP  256
#define MB     64          // tokens per workgroup (4 M-tiles of 16)
#define TOPK   8
#define KC     32          // K per chunk (bf16 WMMA depth)
#define NCHUNK (H_DIM / KC)   // 224
#define AROWB  80          // padded LDS row stride in bytes (40 bf16) -> conflict-free b128 reads
#define ABUFB  (MB * AROWB)   // 5120 B per hi/lo array

// ---------------- prepass: split gate weights into bf16 hi/lo, B-fragment layout ----------------
// layout: element (chunk, col, khalf, kk) at ((chunk*256 + col)*2 + khalf)*16 + kk
// so a lane (col, khalf) reads its 16 K-values as one contiguous 32-byte v16bf.
__global__ __launch_bounds__(256)
void convert_w_kernel(const float* __restrict__ Wg,
                      __bf16* __restrict__ whi, __bf16* __restrict__ wlo)
{
    const int gid   = blockIdx.x * 256 + threadIdx.x;  // 448 blocks * 256 = 224*2*256
    const int col   = gid & 255;
    const int pair  = gid >> 8;          // chunk*2 + khalf
    const int chunk = pair >> 1;
    const int khalf = pair & 1;
    const int krow0 = chunk * KC + khalf * 16;

    union { v16bf v; __bf16 b[16]; } H, L;
#pragma unroll
    for (int kk = 0; kk < 16; ++kk) {
        const float f = Wg[(size_t)(krow0 + kk) * N_EXP + col];
        const __bf16 h = (__bf16)f;              // RTNE high part
        H.b[kk] = h;
        L.b[kk] = (__bf16)(f - (float)h);        // residual
    }
    const size_t base = ((size_t)(chunk * N_EXP + col) * 2 + khalf) * 16;
    *(v16bf*)(whi + base) = H.v;
    *(v16bf*)(wlo + base) = L.v;
}

// ---------------- fused gate GEMM (split-bf16 WMMA) + grouped top-k ----------------
__global__ __launch_bounds__(256)
void moe_gate_kernel(const float* __restrict__ x,
                     const __bf16* __restrict__ whi,
                     const __bf16* __restrict__ wlo,
                     const float* __restrict__ bias,
                     int*   __restrict__ out_idx,
                     float* __restrict__ out_w)
{
    // 64 KB LDS: during the GEMM it holds double-buffered A hi/lo staging
    // (4 * 5120 B = 20 KB); afterwards it is reused as the 64x256 score board.
    // Buffer 'buf' (0/1): hi at smem + buf*2*ABUFB, lo at smem + buf*2*ABUFB + ABUFB.
    __shared__ __align__(16) unsigned char smem[MB * N_EXP * 4];

    const int tid   = threadIdx.x;
    const int wave  = tid >> 5;      // 0..7 -> experts [wave*32, wave*32+32)
    const int lane  = tid & 31;
    const int l15   = lane & 15;
    const int kh    = lane >> 4;     // which K-half of the fragment this lane holds
    const int tok0  = blockIdx.x * MB;
    const int ncol0 = wave * 32;

    v8f acc[4][2];
#pragma unroll
    for (int m = 0; m < 4; ++m)
#pragma unroll
        for (int n = 0; n < 2; ++n) { v8f z = {}; acc[m][n] = z; }

    // ---- cooperative stage of one 64x32 fp32 A chunk -> bf16 hi/lo in LDS ----
    const int srow  = tid >> 2;            // 64 rows, 4 threads per row
    const int kseg  = (tid & 3) * 8;       // 8 consecutive K per thread
    const float* sBase = x + (size_t)(tok0 + srow) * H_DIM + kseg;

    auto stage = [&](int c, int buf) {
        const float* src = sBase + c * KC;
        if ((c & 7) == 0)                        // gfx1250 global_prefetch_b8, ~8 chunks ahead
            __builtin_prefetch(src + 8 * KC, 0, 1);
        const float4 f0 = *(const float4*)(src);
        const float4 f1 = *(const float4*)(src + 4);
        const float ff[8] = { f0.x, f0.y, f0.z, f0.w, f1.x, f1.y, f1.z, f1.w };
        union { uint4 q; __bf16 b[8]; } H, L;
#pragma unroll
        for (int j = 0; j < 8; ++j) {
            const __bf16 h = (__bf16)ff[j];
            H.b[j] = h;
            L.b[j] = (__bf16)(ff[j] - (float)h);
        }
        unsigned char* dH = smem + buf * (2 * ABUFB) + srow * AROWB + kseg * 2;
        *(uint4*)dH           = H.q;            // ds_store_b128 (hi)
        *(uint4*)(dH + ABUFB) = L.q;            // ds_store_b128 (lo)
    };

    // ---- per-chunk compute: 8 global b128 B-feeds + 16 ds b128 A-feeds + 24 WMMA ----
    auto compute = [&](int c, int buf) {
        v16bf bhi[2], blo[2];
#pragma unroll
        for (int n = 0; n < 2; ++n) {
            const int col = ncol0 + n * 16 + l15;
            const size_t base = ((size_t)(c * N_EXP + col) * 2 + kh) * 16;
            bhi[n] = *(const v16bf*)(whi + base);   // 32 B -> 2x global_load_b128
            blo[n] = *(const v16bf*)(wlo + base);
        }
#pragma unroll
        for (int m = 0; m < 4; ++m) {
            const int row = m * 16 + l15;
            // A fragment in this code's contiguous K order (matches B): lanes 0-15 hold K 0-15, lanes 16-31 hold K 16-31
            const unsigned char* pH = smem + buf * (2 * ABUFB) + row * AROWB + kh * 32;
            union { v16bf v; uint4 q[2]; } AH, AL;
            AH.q[0] = *(const uint4*)(pH);              // ds_load_b128
            AH.q[1] = *(const uint4*)(pH + 16);
            AL.q[0] = *(const uint4*)(pH + ABUFB);
            AL.q[1] = *(const uint4*)(pH + ABUFB + 16);
#pragma unroll
            for (int n = 0; n < 2; ++n) {
                acc[m][n] = __builtin_amdgcn_wmma_f32_16x16x32_bf16(
                    false, AH.v, false, bhi[n], (short)0, acc[m][n], false, false);
                acc[m][n] = __builtin_amdgcn_wmma_f32_16x16x32_bf16(
                    false, AH.v, false, blo[n], (short)0, acc[m][n], false, false);
                acc[m][n] = __builtin_amdgcn_wmma_f32_16x16x32_bf16(
                    false, AL.v, false, bhi[n], (short)0, acc[m][n], false, false);
            }
        }
    };

    stage(0, 0);
    for (int c = 0; c < NCHUNK; ++c) {
        __syncthreads();                        // staging of chunk c complete; prior reads drained
        if (c + 1 < NCHUNK) stage(c + 1, (c + 1) & 1);
        compute(c, c & 1);
    }
    __syncthreads();                            // staging buffers dead; reuse LDS as score board

    float* s_scores = (float*)smem;

    // sigmoid + bias, scatter with XOR column swizzle.
    // C/D layout: VGPR i -> M = i + (lane>=16 ? 8 : 0), N = lane&15
#pragma unroll
    for (int m = 0; m < 4; ++m)
#pragma unroll
        for (int n = 0; n < 2; ++n) {
            const int col = ncol0 + n * 16 + l15;
            const float bb = bias[col];
#pragma unroll
            for (int i = 0; i < 8; ++i) {
                const int row = m * 16 + kh * 8 + i;
                const float s = 1.0f / (1.0f + __expf(-acc[m][n][i])) + bb;
                s_scores[row * N_EXP + (col ^ (row & 63))] = s;
            }
        }
    __syncthreads();

    // -------- selection phase: one thread per token --------
    if (tid < MB) {
        const float* row = &s_scores[tid * N_EXP];
        const int swz = tid & 63;

        float gsum[8];
#pragma unroll
        for (int g = 0; g < 8; ++g) {
            float m1 = -1e30f, m2 = -1e30f;
            for (int e = 0; e < 32; ++e) {
                const float v = row[(g * 32 + e) ^ swz];
                if (v > m1) { m2 = m1; m1 = v; }
                else if (v > m2) { m2 = v; }
            }
            gsum[g] = m1 + m2;
        }

        unsigned gmask = 0;
        for (int it = 0; it < 4; ++it) {
            int best = 0; float bv = -1e30f;
            for (int g = 0; g < 8; ++g)
                if (!((gmask >> g) & 1u) && gsum[g] > bv) { bv = gsum[g]; best = g; }
            gmask |= 1u << best;
        }

        float tv[TOPK]; int ti[TOPK];
#pragma unroll
        for (int j = 0; j < TOPK; ++j) { tv[j] = -1e30f; ti[j] = 0; }
        for (int e = 0; e < N_EXP; ++e) {
            if (!((gmask >> (e >> 5)) & 1u)) continue;
            const float v = row[e ^ swz];
            if (v > tv[TOPK - 1]) {
                int j = TOPK - 1;
                while (j > 0 && v > tv[j - 1]) { tv[j] = tv[j - 1]; ti[j] = ti[j - 1]; --j; }
                tv[j] = v; ti[j] = e;
            }
        }

        float den = 1e-20f;
#pragma unroll
        for (int j = 0; j < TOPK; ++j) den += tv[j];
        const float scale = 2.5f / den;

        const int token = tok0 + tid;
#pragma unroll
        for (int j = 0; j < TOPK; ++j) {
            out_idx[token * TOPK + j] = ti[j];
            out_w[token * TOPK + j]   = tv[j] * scale;
        }
    }
}

extern "C" void kernel_launch(void* const* d_in, const int* in_sizes, int n_in,
                              void* d_out, int out_size, void* d_ws, size_t ws_size,
                              hipStream_t stream)
{
    const float* x    = (const float*)d_in[0];  // [4,4096,7168] f32
    const float* Wg   = (const float*)d_in[1];  // [7168,256]    f32
    const float* bias = (const float*)d_in[2];  // [256]         f32

    const int T = in_sizes[0] / H_DIM;          // 16384 tokens

    // workspace: bf16 hi/lo split of the gate weights (2 * 3.67 MB)
    __bf16* whi = (__bf16*)d_ws;
    __bf16* wlo = whi + (size_t)H_DIM * N_EXP;

    // d_out = topk_idx [T,8] (int32 bits) followed by topk_weight [T,8] (f32)
    int*   out_idx = (int*)d_out;
    float* out_w   = (float*)d_out + (size_t)T * TOPK;

    hipLaunchKernelGGL(convert_w_kernel, dim3(NCHUNK * 2), dim3(256), 0, stream,
                       Wg, whi, wlo);
    hipLaunchKernelGGL(moe_gate_kernel, dim3(T / MB), dim3(256), 0, stream,
                       x, whi, wlo, bias, out_idx, out_w);
}
